// EEA_head_76321568850443
// MI455X (gfx1250) — matrix operand order repaired and checked by reading, not compiled, not run on hardware
//
#include <hip/hip_runtime.h>

// ---------------- problem constants ----------------
#define B_    8
#define C_    128
#define H_    88
#define W_    88
#define HWPIX (H_ * W_)          // 7744
#define NPIX  (B_ * HWPIX)       // 61952 total pixels
#define NWD   22                 // windows per spatial dim (88/4)
#define NWIN  (B_ * NWD * NWD)   // 3872 windows
#define TOK   16                 // tokens per window (4x4)
#define HID_  512
#define HID2  1024
#define EPS_  1e-5f
#define SCALE_ 0.125f            // (128/2)^-0.5

typedef __attribute__((ext_vector_type(16))) _Float16 v16h;
typedef __attribute__((ext_vector_type(8)))  _Float16 v8h;
typedef __attribute__((ext_vector_type(4)))  _Float16 v4h;
typedef __attribute__((ext_vector_type(8)))  float    v8f;
typedef __attribute__((ext_vector_type(2)))  float    v2f;

// ---------------- WMMA wrappers ----------------
__device__ __forceinline__ v8f wmma_f16(v16h a, v16h b, v8f c) {
  // D(16x16,f32) = A(16x32,f16) * B(32x16,f16) + C
  return __builtin_amdgcn_wmma_f32_16x16x32_f16(false, a, false, b, (short)0, c,
                                                false, false);
}
__device__ __forceinline__ v8f wmma_f32k4(v2f a, v2f b, v8f c) {
  // D(16x16,f32) = A(16x4,f32) * B(4x16,f32) + C
  return __builtin_amdgcn_wmma_f32_16x16x4_f32(false, a, false, b, (short)0, c,
                                               false, false);
}

// f16 B-matrix (32x16) per-lane K mapping used by the fragment repack:
// lane&15 = N; lanes 0-15 hold K=0..7,16..23, lanes 16-31 hold K=8..15,24..31
__device__ __forceinline__ int kmapB(int s, int lane) {
  return (s < 8 ? s : s + 8) + ((lane >= 16) ? 8 : 0);
}

__device__ __forceinline__ v16h cat16(v8h lo, v8h hi) {
  return __builtin_shufflevector(lo, hi, 0, 1, 2, 3, 4, 5, 6, 7, 8, 9, 10, 11,
                                 12, 13, 14, 15);
}
// A fragment (16x32 f16): per-lane K = {base..base+7} and {base+16..base+23},
// base = kc*32 + 8*(lane>=16). Two aligned b128 loads.
__device__ __forceinline__ v16h loadA(const _Float16* row, int kc, int half) {
  const v8h* p = (const v8h*)(row + kc * 32 + half * 8);
  return cat16(p[0], p[2]);
}
// B fragment from fragment-ready weights: [tile][lane][16 halfs] contiguous.
__device__ __forceinline__ v16h loadB(const _Float16* frag, int tile, int lane) {
  const v8h* p = (const v8h*)(frag + ((size_t)tile * 32 + lane) * 16);
  return cat16(p[0], p[1]);
}

// ---------------- reductions (wave32) ----------------
__device__ __forceinline__ float wsum32(float v) {
#pragma unroll
  for (int m = 16; m >= 1; m >>= 1) v += __shfl_xor(v, m, 32);
  return v;
}
__device__ __forceinline__ float hsum16(float v) {
#pragma unroll
  for (int m = 8; m >= 1; m >>= 1) v += __shfl_xor(v, m, 32);
  return v;
}
__device__ __forceinline__ float hmax16(float v) {
#pragma unroll
  for (int m = 8; m >= 1; m >>= 1) v = fmaxf(v, __shfl_xor(v, m, 32));
  return v;
}

// half-pixel bilinear tap into a 44x44 plane (align_corners=False, scale 2)
__device__ __forceinline__ float bilerp44(const float* p, float sy, float sx) {
  int y0 = (int)floorf(sy), x0 = (int)floorf(sx);
  float wy = sy - (float)y0, wx = sx - (float)x0;
  int y0c = min(max(y0, 0), 43), y1c = min(max(y0 + 1, 0), 43);
  int x0c = min(max(x0, 0), 43), x1c = min(max(x0 + 1, 0), 43);
  float v00 = p[y0c * 44 + x0c], v01 = p[y0c * 44 + x1c];
  float v10 = p[y1c * 44 + x0c], v11 = p[y1c * 44 + x1c];
  return v00 * (1.f - wy) * (1.f - wx) + v01 * (1.f - wy) * wx +
         v10 * wy * (1.f - wx) + v11 * wy * wx;
}

// ====== kernel 0: repack weights fp32 -> fp16 fragment-ready layout ======
// frag[((kc*NT + nt)*32 + lane)*16 + s] = W[(nt*16 + (lane&15))*K + kc*32 + kmapB(s,lane)]
__global__ void k_weights(const float* wq, const float* wk, const float* wv,
                          const float* pin, const float* pout,
                          _Float16* wqf, _Float16* wkf, _Float16* wvf,
                          _Float16* pinf, _Float16* poutf) {
  int gid = blockIdx.x * 256 + threadIdx.x;  // 245760 total
  const float* src;
  _Float16* dst;
  int l, NT, K;
  if (gid < 16384) {
    src = wq; dst = wqf; l = gid; NT = 8; K = 128;
  } else if (gid < 32768) {
    src = wk; dst = wkf; l = gid - 16384; NT = 8; K = 128;
  } else if (gid < 49152) {
    src = wv; dst = wvf; l = gid - 32768; NT = 8; K = 128;
  } else if (gid < 49152 + 131072) {
    src = pin; dst = pinf; l = gid - 49152; NT = 64; K = 128;
  } else {
    src = pout; dst = poutf; l = gid - (49152 + 131072); NT = 8; K = 512;
  }
  int s = l & 15, lane = (l >> 4) & 31, tile = l >> 9;
  int kc = tile / NT, nt = tile % NT;
  int n = nt * 16 + (lane & 15);
  int k = kc * 32 + kmapB(s, lane);
  dst[l] = (_Float16)src[n * K + k];
}

// ========== kernel 1: LN1 + bilinear upsample + windowize (one wave / pixel) ==========
__global__ void k_prep(const float* __restrict__ x, const float* __restrict__ mask,
                       const float* __restrict__ edge, const float* __restrict__ n1w,
                       const float* __restrict__ n1b,
                       _Float16* __restrict__ qwin, _Float16* __restrict__ xkwin) {
  int lane = threadIdx.x & 31, wid = threadIdx.x >> 5;
  int wg = blockIdx.x * 8 + wid;                // pixel id, 0..NPIX-1
  int b = wg / HWPIX, pix = wg % HWPIX;
  int h = pix / W_, w = pix % W_;
  float sy = (float)h * 0.5f - 0.25f, sx = (float)w * 0.5f - 0.25f;

  float xv[4]; float s = 0.f, s2 = 0.f;
#pragma unroll
  for (int u = 0; u < 4; ++u) {
    int c = lane + 32 * u;
    float v = x[((size_t)(b * C_ + c)) * HWPIX + pix];
    xv[u] = v; s += v; s2 += v * v;
  }
  s = wsum32(s); s2 = wsum32(s2);
  float mu = s * (1.f / 128.f);
  float var = s2 * (1.f / 128.f) - mu * mu;
  float rstd = rsqrtf(var + EPS_);

  int win = (b * NWD + h / 4) * NWD + w / 4;
  int t = (h & 3) * 4 + (w & 3);
  size_t base = (size_t)win * (TOK * C_) + (size_t)t * C_;
#pragma unroll
  for (int u = 0; u < 4; ++u) {
    int c = lane + 32 * u;
    const float* mp = mask + ((size_t)(b * C_ + c)) * (44 * 44);
    const float* ep = edge + ((size_t)(b * C_ + c)) * (44 * 44);
    float mk = bilerp44(mp, sy, sx);
    float eg = bilerp44(ep, sy, sx);
    float xln = (xv[u] - mu) * rstd * n1w[c] + n1b[c];
    xkwin[base + c] = (_Float16)(xln * mk);
    qwin[base + c] = (_Float16)eg;
  }
}

// ========== kernel 2: per-window attention (one wave / window) ==========
__global__ void k_attn(const _Float16* __restrict__ qwin,
                       const _Float16* __restrict__ xkwin,
                       const _Float16* __restrict__ wqf,
                       const _Float16* __restrict__ wkf,
                       const _Float16* __restrict__ wvf,
                       float* __restrict__ outwin) {
  __shared__ _Float16 sm[4][3][TOK][C_];   // 48 KB: q/k/v tiles per wave
  int lane = threadIdx.x & 31, wid = threadIdx.x >> 5;
  int win = blockIdx.x * 4 + wid;
  int ln = lane & 15, half = (lane >= 16) ? 1 : 0;

  // ---- projections: D = A(16x128) * W^T(128x128), K chunks of 32 ----
  for (int proj = 0; proj < 3; ++proj) {
    const _Float16* src = (proj == 0) ? qwin : xkwin;
    const _Float16* Wf = (proj == 0) ? wqf : ((proj == 1) ? wkf : wvf);
    const _Float16* arow = src + (size_t)win * (TOK * C_) + ln * C_;
    v8f acc[8];
#pragma unroll
    for (int nt = 0; nt < 8; ++nt) acc[nt] = (v8f){};
#pragma unroll
    for (int kc = 0; kc < 4; ++kc) {
      v16h a = loadA(arow, kc, half);
      if (kc < 3) __builtin_prefetch(arow + (kc + 1) * 32, 0, 1);
#pragma unroll
      for (int nt = 0; nt < 8; ++nt) {
        v16h bf = loadB(Wf, kc * 8 + nt, lane);
        acc[nt] = wmma_f16(a, bf, acc[nt]);
      }
    }
    // D layout: M = r + 8*half (token), N = ln (channel-in-tile)
#pragma unroll
    for (int nt = 0; nt < 8; ++nt)
#pragma unroll
      for (int r = 0; r < 8; ++r)
        sm[wid][proj][r + 8 * half][nt * 16 + ln] = (_Float16)acc[nt][r];
  }

  const _Float16* Q = &sm[wid][0][0][0];
  const _Float16* K = &sm[wid][1][0][0];
  const _Float16* V = &sm[wid][2][0][0];

  // ---- per-token: dots(64x64) = Qt(64x2)*Kt(2x64), softmax, out(2x64) ----
  for (int t = 0; t < TOK; ++t) {
    v8f accD[4][4];
#pragma unroll
    for (int jt = 0; jt < 4; ++jt) {
      v2f aF;
#pragma unroll
      for (int r = 0; r < 2; ++r) {
        int ki = r + 2 * half;     // head index, pad K 2->4 with zeros
        aF[r] = (ki < 2) ? (float)Q[t * C_ + ki * 64 + jt * 16 + ln] : 0.f;
      }
#pragma unroll
      for (int kt = 0; kt < 4; ++kt) {
        v2f bF;
#pragma unroll
        for (int r = 0; r < 2; ++r) {
          int ki = r + 2 * half;
          bF[r] = (ki < 2) ? (float)K[t * C_ + ki * 64 + kt * 16 + ln] : 0.f;
        }
        v8f cz = (v8f){};
        accD[jt][kt] = wmma_f32k4(aF, bF, cz);
      }
    }
    // softmax over k (4 kt tiles x 16 lanes of this half), rows = (jt, r, half)
#pragma unroll
    for (int jt = 0; jt < 4; ++jt) {
#pragma unroll
      for (int r = 0; r < 8; ++r) {
        float mx = -3.4e38f;
#pragma unroll
        for (int kt = 0; kt < 4; ++kt) mx = fmaxf(mx, accD[jt][kt][r] * SCALE_);
        mx = hmax16(mx);
        float e[4], sum = 0.f;
#pragma unroll
        for (int kt = 0; kt < 4; ++kt) {
          e[kt] = __expf(accD[jt][kt][r] * SCALE_ - mx);
          sum += e[kt];
        }
        sum = hsum16(sum);
        float inv = 1.f / sum;
#pragma unroll
        for (int kt = 0; kt < 4; ++kt) accD[jt][kt][r] = e[kt] * inv;
      }
    }
    // out[t,i,j] = sum_k attn[j,k] * v[t,i,k]
#pragma unroll
    for (int i = 0; i < 2; ++i) {
#pragma unroll
      for (int jt = 0; jt < 4; ++jt) {
#pragma unroll
        for (int r = 0; r < 8; ++r) {
          float sacc = 0.f;
#pragma unroll
          for (int kt = 0; kt < 4; ++kt)
            sacc += accD[jt][kt][r] * (float)V[t * C_ + i * 64 + kt * 16 + ln];
          sacc = hsum16(sacc);
          if (ln == 0) {
            int j = jt * 16 + r + 8 * half;
            outwin[(size_t)win * (TOK * C_) + t * C_ + i * 64 + j] = sacc;
          }
        }
      }
    }
  }
}

// ========== kernel 3: scrambled reassembly + residual + LN2 (one wave / pixel) ==========
__global__ void k_reasm(const float* __restrict__ x, const float* __restrict__ outwin,
                        const float* __restrict__ n2w, const float* __restrict__ n2b,
                        float* __restrict__ y1, _Float16* __restrict__ ln2) {
  int lane = threadIdx.x & 31, wid = threadIdx.x >> 5;
  int wg = blockIdx.x * 8 + wid;
  int b = wg / HWPIX, pix = wg % HWPIX;

  float yv[4]; float s = 0.f, s2 = 0.f;
#pragma unroll
  for (int u = 0; u < 4; ++u) {
    int c = lane + 32 * u;
    // flat (c,h,w) index -> (p, i, j) of (2048, 22, 22) view
    int F = c * HWPIX + pix;
    int p = F / 484, rem = F - p * 484;
    int i = rem / 22, j = rem % 22;
    int win = (b * NWD + i) * NWD + j;
    int t = p >> 7, cc = p & 127;
    float att = outwin[(size_t)win * (TOK * C_) + t * C_ + cc];
    float v = x[((size_t)(b * C_ + c)) * HWPIX + pix] + att;
    yv[u] = v; s += v; s2 += v * v;
    y1[((size_t)(b * C_ + c)) * HWPIX + pix] = v;
  }
  s = wsum32(s); s2 = wsum32(s2);
  float mu = s * (1.f / 128.f);
  float var = s2 * (1.f / 128.f) - mu * mu;
  float rstd = rsqrtf(var + EPS_);
#pragma unroll
  for (int u = 0; u < 4; ++u) {
    int c = lane + 32 * u;
    ln2[((size_t)(b * HWPIX + pix)) * C_ + c] =
        (_Float16)((yv[u] - mu) * rstd * n2w[c] + n2b[c]);
  }
}

// ========== kernel 4: FFN project_in GEMM (M=61952, K=128, N=1024) ==========
__global__ void k_pin(const _Float16* __restrict__ ln2, const _Float16* __restrict__ pinf,
                      _Float16* __restrict__ hbuf) {
  int lane = threadIdx.x & 31, wid = threadIdx.x >> 5;
  int wg = blockIdx.x * 8 + wid;          // 30976 waves
  int mt = wg >> 3, ng = wg & 7;          // M-tile, N-group (128 cols)
  int ln = lane & 15, half = (lane >= 16) ? 1 : 0;
  int row0 = mt * 16;
  const _Float16* arow = ln2 + (size_t)(row0 + ln) * C_;

  v8f acc[8];
#pragma unroll
  for (int nt = 0; nt < 8; ++nt) acc[nt] = (v8f){};
#pragma unroll
  for (int kc = 0; kc < 4; ++kc) {
    v16h a = loadA(arow, kc, half);
    if (kc < 3) __builtin_prefetch(arow + (kc + 1) * 32, 0, 1);
#pragma unroll
    for (int nt = 0; nt < 8; ++nt) {
      v16h bf = loadB(pinf, kc * 64 + ng * 8 + nt, lane);
      acc[nt] = wmma_f16(a, bf, acc[nt]);
    }
  }
#pragma unroll
  for (int nt = 0; nt < 8; ++nt)
#pragma unroll
    for (int r = 0; r < 8; ++r)
      hbuf[(size_t)(row0 + r + 8 * half) * HID2 + ng * 128 + nt * 16 + ln] =
          (_Float16)acc[nt][r];
}

// ========== kernel 5: depthwise 3x3 + GELU gate (4 channels / thread) ==========
__global__ void k_dw(const _Float16* __restrict__ hbuf, const float* __restrict__ dww,
                     _Float16* __restrict__ gbuf) {
  int gid = blockIdx.x * 256 + threadIdx.x;   // NPIX * 128 threads
  int c0 = (gid & 127) * 4;                   // 4 consecutive gate channels
  int rest = gid >> 7;
  int pix = rest % HWPIX, b = rest / HWPIX;
  int py = pix / W_, px = pix % W_;

  float x1[4] = {0.f, 0.f, 0.f, 0.f}, x2[4] = {0.f, 0.f, 0.f, 0.f};
#pragma unroll
  for (int dy = 0; dy < 3; ++dy) {
    int ny = py + dy - 1;
    if (ny < 0 || ny >= H_) continue;
#pragma unroll
    for (int dx = 0; dx < 3; ++dx) {
      int nx = px + dx - 1;
      if (nx < 0 || nx >= W_) continue;
      size_t nb = (size_t)(b * HWPIX + ny * W_ + nx) * HID2;
      v4h h1 = *(const v4h*)(hbuf + nb + c0);         // x1 half
      v4h h2 = *(const v4h*)(hbuf + nb + c0 + 512);   // x2 half
#pragma unroll
      for (int u = 0; u < 4; ++u) {
        x1[u] += (float)h1[u] * dww[(c0 + u) * 9 + dy * 3 + dx];
        x2[u] += (float)h2[u] * dww[(c0 + u + 512) * 9 + dy * 3 + dx];
      }
    }
  }
  v4h g4;
#pragma unroll
  for (int u = 0; u < 4; ++u) {
    float g = 0.5f * x1[u] * (1.f + erff(x1[u] * 0.70710678118f)); // exact GELU
    g4[u] = (_Float16)(g * x2[u]);
  }
  *(v4h*)(gbuf + (size_t)(b * HWPIX + pix) * HID_ + c0) = g4;
}

// ========== kernel 6: FFN project_out GEMM (K=512, N=128) + residual ==========
__global__ void k_pout(const _Float16* __restrict__ gbuf, const _Float16* __restrict__ poutf,
                       const float* __restrict__ y1, float* __restrict__ out) {
  int lane = threadIdx.x & 31, wid = threadIdx.x >> 5;
  int mt = blockIdx.x * 8 + wid;          // 3872 M-tiles
  int ln = lane & 15, half = (lane >= 16) ? 1 : 0;
  int row0 = mt * 16;
  const _Float16* arow = gbuf + (size_t)(row0 + ln) * HID_;

  v8f acc[8];
#pragma unroll
  for (int nt = 0; nt < 8; ++nt) acc[nt] = (v8f){};
#pragma unroll
  for (int kc = 0; kc < 16; ++kc) {       // 512 / 32
    v16h a = loadA(arow, kc, half);
    if (kc < 15) __builtin_prefetch(arow + (kc + 1) * 32, 0, 1);
#pragma unroll
    for (int nt = 0; nt < 8; ++nt) {
      v16h bf = loadB(poutf, kc * 8 + nt, lane);
      acc[nt] = wmma_f16(a, bf, acc[nt]);
    }
  }
#pragma unroll
  for (int nt = 0; nt < 8; ++nt)
#pragma unroll
    for (int r = 0; r < 8; ++r) {
      int row = row0 + r + 8 * half;
      int b = row / HWPIX, pix = row % HWPIX;
      int c = nt * 16 + ln;
      size_t idx = ((size_t)(b * C_ + c)) * HWPIX + pix;
      out[idx] = y1[idx] + acc[nt][r];
    }
}

// ================= host-side launcher =================
extern "C" void kernel_launch(void* const* d_in, const int* in_sizes, int n_in,
                              void* d_out, int out_size, void* d_ws, size_t ws_size,
                              hipStream_t stream) {
  const float* x    = (const float*)d_in[0];
  const float* mask = (const float*)d_in[1];
  const float* edge = (const float*)d_in[2];
  const float* wq   = (const float*)d_in[3];
  const float* wk   = (const float*)d_in[4];
  const float* wv   = (const float*)d_in[5];
  const float* n1w  = (const float*)d_in[6];
  const float* n1b  = (const float*)d_in[7];
  const float* n2w  = (const float*)d_in[8];
  const float* n2b  = (const float*)d_in[9];
  const float* pin  = (const float*)d_in[10];
  const float* dww  = (const float*)d_in[11];
  const float* pout = (const float*)d_in[12];
  float* outp = (float*)d_out;

  char* ws = (char*)d_ws;
  auto carve = [&](size_t bytes) -> char* {
    char* p = ws;
    ws += (bytes + 255) & ~(size_t)255;
    return p;
  };
  _Float16* qwin  = (_Float16*)carve((size_t)NWIN * TOK * C_ * 2);
  _Float16* xkwin = (_Float16*)carve((size_t)NWIN * TOK * C_ * 2);
  _Float16* wqf   = (_Float16*)carve(16384 * 2);
  _Float16* wkf   = (_Float16*)carve(16384 * 2);
  _Float16* wvf   = (_Float16*)carve(16384 * 2);
  _Float16* pinf  = (_Float16*)carve(128 * 1024 * 2);
  _Float16* poutf = (_Float16*)carve(512 * 128 * 2);
  float*    outwin= (float*)carve((size_t)NWIN * TOK * C_ * 4);
  float*    y1    = (float*)carve((size_t)NPIX * C_ * 4);
  _Float16* ln2   = (_Float16*)carve((size_t)NPIX * C_ * 2);
  _Float16* hbuf  = (_Float16*)carve((size_t)NPIX * HID2 * 2);
  _Float16* gbuf  = (_Float16*)carve((size_t)NPIX * HID_ * 2);

  k_weights<<<960, 256, 0, stream>>>(wq, wk, wv, pin, pout, wqf, wkf, wvf, pinf, poutf);
  k_prep<<<NPIX / 8, 256, 0, stream>>>(x, mask, edge, n1w, n1b, qwin, xkwin);
  k_attn<<<NWIN / 4, 128, 0, stream>>>(qwin, xkwin, wqf, wkf, wvf, outwin);
  k_reasm<<<NPIX / 8, 256, 0, stream>>>(x, outwin, n2w, n2b, y1, ln2);
  k_pin<<<NPIX / 16, 256, 0, stream>>>(ln2, pinf, hbuf);
  k_dw<<<(NPIX * 128) / 256, 256, 0, stream>>>(hbuf, dww, gbuf);
  k_pout<<<(NPIX / 16) / 8, 256, 0, stream>>>(gbuf, poutf, y1, outp);
}
